// Deformation_81071802679462
// MI455X (gfx1250) — matrix operand order repaired and checked by reading, not compiled, not run on hardware
//
#include <hip/hip_runtime.h>

#define FDIM   64
#define WDIM   256
#define RESX   128
#define TRESX  64
#define BOUNDS 1.3f
#define TILE_M 64

typedef __bf16 bf16_t;
typedef __attribute__((ext_vector_type(16))) __bf16 v16bf;
typedef __attribute__((ext_vector_type(8)))  __bf16 v8bf;
typedef __attribute__((ext_vector_type(8)))  float  v8f;
typedef __attribute__((ext_vector_type(4)))  int    v4i;

#if defined(__HIP_DEVICE_COMPILE__) && __has_builtin(__builtin_amdgcn_global_load_async_to_lds_b128)
#define HAVE_ASYNC_LDS 1
#endif

#define ASG __attribute__((address_space(1)))
#define ASL __attribute__((address_space(3)))

// LDS strides in halves (padded to break bank conflicts, keep 16B alignment)
#define WLD 264   // transposed-weight rows: [n][k]
#define SLD 264   // activation rows: [m][k], 64 x 256
#define FLD 72    // feature rows: 64 x 64

// LDS carve (bytes)
#define OFF_W    0
#define OFF_ST   135168                    // 256*WLD*2
#define OFF_SB   (OFF_ST + 33792)          // 64*SLD*2
#define OFF_H    (OFF_SB + 33792)
#define OFF_FT   (OFF_H  + 33792)
#define OFF_FB   (OFF_FT + 9216)           // 64*FLD*2
#define OFF_POS  (OFF_FB + 9216)
#define OFF_BG   (OFF_POS + 4096)          // 64*16*4
#define OFF_ROT  (OFF_BG  + 4096)
#define OFF_SHS  (OFF_ROT + 4096)
#define OFF_W2   (OFF_SHS + 12288)         // 48*WLD*2 = 25344
#define SMEM_TOTAL (OFF_W2 + 25344)        // 304896 B (<= 320KB/WGP)

// Workspace layout (offsets in bf16 halves): pre-transposed bf16 weight panels
#define WS_ST    0                         // 256 x 64
#define WS_BST   16384                     // 256 x 64
#define WS_W1(h) (32768 + (h) * 65536)     // 256 x 256 each
#define WS_W2(h) (294912 + (h) * 12288)    // 48 x 256 each (zero-padded rows)

union V16U { v16bf v; struct { v8bf lo; v8bf hi; } s; };

__device__ __forceinline__ v8f wmma_bf16(v16bf a, v16bf b, v8f c) {
    return __builtin_amdgcn_wmma_f32_16x16x32_bf16(false, a, false, b, (short)0, c,
                                                   false, false);
}

// A fragment (16x32 bf16): lane L<16 -> row m0+L, halves [k0..k0+7] & [k0+16..k0+23]
//                          lane L>=16 -> row m0+L-16, halves [k0+8..k0+15] & [k0+24..k0+31]
__device__ __forceinline__ v16bf load_a_frag(const bf16_t* buf, int ld, int m0, int k0) {
    unsigned lane = threadIdx.x & 31u;
    int row = m0 + (int)(lane & 15u);
    int kk  = k0 + (int)((lane >> 4) << 3);
    V16U r;
    r.s.lo = *(const v8bf*)(buf + row * ld + kk);
    r.s.hi = *(const v8bf*)(buf + row * ld + kk + 16);
    return r.v;
}

// B fragment (32x16 bf16) from transposed weights WT[n][k]
__device__ __forceinline__ v16bf load_b_frag(const bf16_t* wt, int ld, int n0, int k0) {
    unsigned lane = threadIdx.x & 31u;
    int row = n0 + (int)(lane & 15u);
    int kk  = k0 + (int)((lane >> 4) << 3);
    V16U r;
    r.s.lo = *(const v8bf*)(wt + row * ld + kk);
    r.s.hi = *(const v8bf*)(wt + row * ld + kk + 16);
    return r.v;
}

// Issue async copy of pre-transposed bf16 panel (packed rows of K halves) into LDS.
__device__ __forceinline__ void stage_issue(bf16_t* dst, const bf16_t* __restrict__ src,
                                            int rows, int K) {
    int cpr = K >> 3;                      // 16B chunks per row
    int total = rows * cpr;
    for (int i = threadIdx.x; i < total; i += 256) {
        int r = i / cpr, c = (i - r * cpr) << 3;
#ifdef HAVE_ASYNC_LDS
        __builtin_amdgcn_global_load_async_to_lds_b128(
            (ASG v4i*)(src + r * K + c), (ASL v4i*)(dst + r * WLD + c), 0, 0);
#else
        *(v8bf*)(dst + r * WLD + c) = *(const v8bf*)(src + r * K + c);
#endif
    }
}

__device__ __forceinline__ void async_join() {
#ifdef HAVE_ASYNC_LDS
#if __has_builtin(__builtin_amdgcn_s_wait_asynccnt)
    __builtin_amdgcn_s_wait_asynccnt(0);
#else
    asm volatile("s_wait_asynccnt 0" ::: "memory");
#endif
#endif
}

// D = relu(A @ W + bias) -> bf16 activation buffer. mt fixed per wave; A hoisted.
template <int KTILES>
__device__ void gemm_bias_relu_t(const bf16_t* A, int ldA, const bf16_t* WT, int ldW,
                                 const float* __restrict__ bias, bf16_t* out, int ldO,
                                 int Ntiles) {
    int wave = threadIdx.x >> 5;
    unsigned lane = threadIdx.x & 31u;
    int mt = wave & 3;
    v16bf af[KTILES];
#pragma unroll
    for (int k = 0; k < KTILES; ++k) af[k] = load_a_frag(A, ldA, mt * 16, k * 32);
    for (int nt = wave >> 2; nt < Ntiles; nt += 2) {
        v8f acc = {0.f, 0.f, 0.f, 0.f, 0.f, 0.f, 0.f, 0.f};
#pragma unroll
        for (int k = 0; k < KTILES; ++k) {
            v16bf b = load_b_frag(WT, ldW, nt * 16, k * 32);
            acc = wmma_bf16(af[k], b, acc);
        }
        int nn = nt * 16 + (int)(lane & 15u);
        float bv = bias[nn];
        int mbase = mt * 16 + (int)((lane >> 4) << 3);
#pragma unroll
        for (int r = 0; r < 8; ++r)
            out[(mbase + r) * ldO + nn] = (bf16_t)fmaxf(acc[r] + bv, 0.0f);
    }
}

// D = A @ W + bias -> fp32 staging buffer
template <int KTILES>
__device__ void gemm_bias_f32_t(const bf16_t* A, int ldA, const bf16_t* WT, int ldW,
                                const float* __restrict__ bias, int nValid,
                                float* out, int ldO, int Ntiles) {
    int wave = threadIdx.x >> 5;
    unsigned lane = threadIdx.x & 31u;
    int mt = wave & 3;
    v16bf af[KTILES];
#pragma unroll
    for (int k = 0; k < KTILES; ++k) af[k] = load_a_frag(A, ldA, mt * 16, k * 32);
    for (int nt = wave >> 2; nt < Ntiles; nt += 2) {
        v8f acc = {0.f, 0.f, 0.f, 0.f, 0.f, 0.f, 0.f, 0.f};
#pragma unroll
        for (int k = 0; k < KTILES; ++k) {
            v16bf b = load_b_frag(WT, ldW, nt * 16, k * 32);
            acc = wmma_bf16(af[k], b, acc);
        }
        int nn = nt * 16 + (int)(lane & 15u);
        float bv = (nn < nValid) ? bias[nn] : 0.0f;
        int mbase = mt * 16 + (int)((lane >> 4) << 3);
#pragma unroll
        for (int r = 0; r < 8; ++r)
            out[(mbase + r) * ldO + nn] = acc[r] + bv;
    }
}

__device__ __forceinline__ float bil(const float* __restrict__ pl, int R1, int R2,
                                     float u, float v, int c) {
    u = fminf(fmaxf(u, 0.0f), (float)(R1 - 1));
    v = fminf(fmaxf(v, 0.0f), (float)(R2 - 1));
    int u0 = (int)u, v0 = (int)v;
    int u1 = min(u0 + 1, R1 - 1), v1 = min(v0 + 1, R2 - 1);
    float fu = u - (float)u0, fv = v - (float)v0;
    float p00 = pl[((u0 * R2 + v0) << 6) + c];
    float p10 = pl[((u1 * R2 + v0) << 6) + c];
    float p01 = pl[((u0 * R2 + v1) << 6) + c];
    float p11 = pl[((u1 * R2 + v1) << 6) + c];
    return p00 * (1.f - fu) * (1.f - fv) + p10 * fu * (1.f - fv) +
           p01 * (1.f - fu) * fv + p11 * fu * fv;
}

__device__ __forceinline__ float grid_feat(const float* __restrict__ sp,
                                           const float* __restrict__ tp,
                                           float ux, float uy, float uz, float tt, int c) {
    const int SP = RESX * RESX * FDIM, TP = RESX * TRESX * FDIM;
    float s = bil(sp, RESX, RESX, ux, uy, c) *
              bil(sp + SP, RESX, RESX, ux, uz, c) *
              bil(sp + 2 * SP, RESX, RESX, uy, uz, c);
    float st = bil(tp, RESX, TRESX, ux, tt, c) *
               bil(tp + TP, RESX, TRESX, uy, tt, c) *
               bil(tp + 2 * TP, RESX, TRESX, uz, tt, c);
    return s * st;
}

__device__ __forceinline__ void head_pass(const bf16_t* S, bf16_t* Wbuf, bf16_t* W2buf,
                                          bf16_t* Hbuf,
                                          const bf16_t* w1T, const float* b1,
                                          const bf16_t* w2T, const float* b2,
                                          float* outbuf, int outd, int ntiles) {
    stage_issue(Wbuf, w1T, WDIM, WDIM);        // big layer-1 panel
    stage_issue(W2buf, w2T, ntiles * 16, WDIM); // small layer-2 panel rides along
    async_join();
    __syncthreads();
    gemm_bias_relu_t<8>(S, SLD, Wbuf, WLD, b1, Hbuf, SLD, WDIM / 16);
    __syncthreads();
    gemm_bias_f32_t<8>(Hbuf, SLD, W2buf, WLD, b2, outd, outbuf, ntiles * 16, ntiles);
    __syncthreads();
}

// One-time (per launch) weight prep: W (K x Nw fp32) -> WT[n][k] packed bf16, zero pad.
__global__ __launch_bounds__(256) void prep_wT_kernel(const float* __restrict__ w,
                                                      bf16_t* __restrict__ dst,
                                                      int K, int Nw, int padN) {
    int i = blockIdx.x * blockDim.x + threadIdx.x;
    if (i < padN * K) {
        int nn = i / K, k = i - nn * K;
        dst[i] = (nn < Nw) ? (bf16_t)w[k * Nw + nn] : (bf16_t)0.0f;
    }
}

__global__ __launch_bounds__(256) void deform_fused_kernel(
    const float* __restrict__ pts_in, const float* __restrict__ rot_in,
    const float* __restrict__ shs_in, const float* __restrict__ time_in,
    const float* __restrict__ h_in, const unsigned char* __restrict__ mask_in,
    const float* __restrict__ tgt_sp, const float* __restrict__ tgt_tp,
    const float* __restrict__ bg_sp, const float* __restrict__ bg_tp,
    const bf16_t* __restrict__ ws,
    const float* __restrict__ st_bb, const float* __restrict__ bst_b,
    const float* __restrict__ pos_b1, const float* __restrict__ pos_b2,
    const float* __restrict__ bpos_b1, const float* __restrict__ bpos_b2,
    const float* __restrict__ rot_b1, const float* __restrict__ rot_b2,
    const float* __restrict__ shs_b1, const float* __restrict__ shs_b2,
    float* __restrict__ out, int n) {
    extern __shared__ __align__(16) char smem[];
    bf16_t* Wbuf  = (bf16_t*)(smem + OFF_W);
    bf16_t* Sbt   = (bf16_t*)(smem + OFF_ST);
    bf16_t* Sbb   = (bf16_t*)(smem + OFF_SB);
    bf16_t* Hbuf  = (bf16_t*)(smem + OFF_H);
    bf16_t* Fbt   = (bf16_t*)(smem + OFF_FT);
    bf16_t* Fbb   = (bf16_t*)(smem + OFF_FB);
    float*  PosB  = (float*)(smem + OFF_POS);
    float*  BgB   = (float*)(smem + OFF_BG);
    float*  RotB  = (float*)(smem + OFF_ROT);
    float*  ShsB  = (float*)(smem + OFF_SHS);
    bf16_t* W2buf = (bf16_t*)(smem + OFF_W2);

    const int pbase = blockIdx.x * TILE_M;

    // Issue both st-panel copies now (tgt at k 0..63, bg at k 64..127 of Wbuf);
    // their latency hides behind the plane-gather phase below.
    stage_issue(Wbuf, ws + WS_ST, WDIM, FDIM);
    stage_issue(Wbuf + FDIM, ws + WS_BST, WDIM, FDIM);

    // ---- Phase A: plane features -> LDS (bf16, A layout) ----
    {
        int c = threadIdx.x & 63;
        int q = threadIdx.x >> 6;
        for (int pi = q; pi < TILE_M; pi += 4) {
            int p = pbase + pi;
            if (p < n) {
                float x = pts_in[p * 3 + 0], y = pts_in[p * 3 + 1], z = pts_in[p * 3 + 2];
                float tv = fminf(fmaxf(time_in[p], 0.0f), 1.0f);
                float ux = fminf(fmaxf((x / BOUNDS + 1.0f) * 0.5f, 0.0f), 1.0f) * 127.0f;
                float uy = fminf(fmaxf((y / BOUNDS + 1.0f) * 0.5f, 0.0f), 1.0f) * 127.0f;
                float uz = fminf(fmaxf((z / BOUNDS + 1.0f) * 0.5f, 0.0f), 1.0f) * 127.0f;
                float tt = tv * 63.0f;
                Fbt[pi * FLD + c] = (bf16_t)grid_feat(tgt_sp, tgt_tp, ux, uy, uz, tt, c);
                Fbb[pi * FLD + c] = (bf16_t)grid_feat(bg_sp, bg_tp, ux, uy, uz, tt, c);
            }
        }
    }
    async_join();
    __syncthreads();

    // ---- GEMM1: relu(feat @ st_w + st_b) for tgt / bg (no barrier needed between) ----
    gemm_bias_relu_t<2>(Fbt, FLD, Wbuf, WLD, st_bb, Sbt, SLD, WDIM / 16);
    gemm_bias_relu_t<2>(Fbb, FLD, Wbuf + FDIM, WLD, bst_b, Sbb, SLD, WDIM / 16);
    __syncthreads();

    // ---- Heads ----
    head_pass(Sbt, Wbuf, W2buf, Hbuf, ws + WS_W1(0), pos_b1, ws + WS_W2(0), pos_b2, PosB, 3, 1);
    head_pass(Sbb, Wbuf, W2buf, Hbuf, ws + WS_W1(1), bpos_b1, ws + WS_W2(1), bpos_b2, BgB, 3, 1);
    head_pass(Sbt, Wbuf, W2buf, Hbuf, ws + WS_W1(2), rot_b1, ws + WS_W2(2), rot_b2, RotB, 4, 1);
    head_pass(Sbt, Wbuf, W2buf, Hbuf, ws + WS_W1(3), shs_b1, ws + WS_W2(3), shs_b2, ShsB, 48, 3);

    // ---- Epilogue ----
    {
        float* out_pts  = out;
        float* out_rot  = out + (size_t)3 * n;
        float* out_opac = out + (size_t)7 * n;
        float* out_shs  = out + (size_t)8 * n;
        int pi = threadIdx.x & 63;
        int rep = threadIdx.x >> 6;
        int p = pbase + pi;
        if (p < n) {
            bool m = mask_in[p] != 0;
            float mf = m ? 1.0f : 0.0f;
            if (rep == 0) {
#pragma unroll
                for (int j = 0; j < 3; ++j) {
                    float d = m ? PosB[pi * 16 + j] : BgB[pi * 16 + j];
                    out_pts[p * 3 + j] = pts_in[p * 3 + j] + d;
                }
                float h0 = h_in[p * 3 + 0], h1 = h_in[p * 3 + 1], h2 = h_in[p * 3 + 2];
                float t0 = time_in[0];
                float wv = h1 * h1;
                float mu = 1.0f / (1.0f + __expf(-h2));
                float op = m ? __expf(-wv * (t0 - mu) * (t0 - mu))
                             : 1.0f / (1.0f + __expf(-h0));
                out_opac[p] = op;
            } else if (rep == 1) {
#pragma unroll
                for (int j = 0; j < 4; ++j)
                    out_rot[p * 4 + j] = rot_in[p * 4 + j] + mf * RotB[pi * 16 + j];
            } else {
                int j0 = (rep - 2) * 24;
#pragma unroll
                for (int j = 0; j < 24; ++j)
                    out_shs[(size_t)p * 48 + j0 + j] =
                        shs_in[(size_t)p * 48 + j0 + j] + mf * ShsB[pi * 48 + j0 + j];
            }
        }
    }
}

extern "C" void kernel_launch(void* const* d_in, const int* in_sizes, int n_in,
                              void* d_out, int out_size, void* d_ws, size_t ws_size,
                              hipStream_t stream) {
    (void)n_in; (void)out_size; (void)ws_size;
    int n = in_sizes[0] / 3;
    bf16_t* ws = (bf16_t*)d_ws;

    // Prep: transpose + convert all weight matrices to bf16 panels in workspace.
    prep_wT_kernel<<<64, 256, 0, stream>>>((const float*)d_in[12], ws + WS_ST, FDIM, WDIM, WDIM);
    prep_wT_kernel<<<64, 256, 0, stream>>>((const float*)d_in[14], ws + WS_BST, FDIM, WDIM, WDIM);
    const int w1i[4] = {16, 20, 24, 28};
    const int w2i[4] = {18, 22, 26, 30};
    const int outd[4] = {3, 3, 4, 48};
    for (int h = 0; h < 4; ++h) {
        prep_wT_kernel<<<256, 256, 0, stream>>>((const float*)d_in[w1i[h]], ws + WS_W1(h),
                                                WDIM, WDIM, WDIM);
        prep_wT_kernel<<<48, 256, 0, stream>>>((const float*)d_in[w2i[h]], ws + WS_W2(h),
                                               WDIM, outd[h], 48);
    }

    dim3 grid((n + TILE_M - 1) / TILE_M), block(256);
    deform_fused_kernel<<<grid, block, SMEM_TOTAL, stream>>>(
        (const float*)d_in[0], (const float*)d_in[1], (const float*)d_in[3],
        (const float*)d_in[5], (const float*)d_in[6], (const unsigned char*)d_in[7],
        (const float*)d_in[8], (const float*)d_in[9], (const float*)d_in[10],
        (const float*)d_in[11], ws,
        (const float*)d_in[13], (const float*)d_in[15],
        (const float*)d_in[17], (const float*)d_in[19],
        (const float*)d_in[21], (const float*)d_in[23],
        (const float*)d_in[25], (const float*)d_in[27],
        (const float*)d_in[29], (const float*)d_in[31],
        (float*)d_out, n);
}
